// GATv2_61254823576188
// MI455X (gfx1250) — matrix operand order repaired and checked by reading, not compiled, not run on hardware
//
#include <hip/hip_runtime.h>
#include <hip/hip_bf16.h>
#include <hip/hip_fp16.h>

// ---------------- problem constants (match reference) ----------------
#define NN      20000
#define E_RAW   200000
#define NEDGE   (E_RAW + NN)   // + self loops
#define NBATCH  512
#define NH      8
#define NC      64
#define HC      512            // H*C
#define F0      64
#define NEG_SLOPE 0.2f

typedef __attribute__((ext_vector_type(16))) _Float16 v16h;
typedef __attribute__((ext_vector_type(8)))  _Float16 v8h;
typedef __attribute__((ext_vector_type(8)))  float    v8f;
typedef unsigned int v4u __attribute__((ext_vector_type(4)));
typedef int          v8i __attribute__((ext_vector_type(8)));
typedef int          v4i __attribute__((ext_vector_type(4)));

#if __has_builtin(__builtin_amdgcn_tensor_load_to_lds)
#define USE_TDM 1
#else
#define USE_TDM 0
#endif

// order-preserving float<->uint encoding for atomicMax-based segment max
__device__ __forceinline__ unsigned enc_f32(float f) {
  unsigned u = __float_as_uint(f);
  return (u & 0x80000000u) ? ~u : (u | 0x80000000u);
}
__device__ __forceinline__ float dec_f32(unsigned k) {
  unsigned u = (k & 0x80000000u) ? (k ^ 0x80000000u) : ~k;
  return __uint_as_float(u);
}

__device__ __forceinline__ int esrc(const int* ei, int e) { return e < E_RAW ? ei[e]         : (e - E_RAW); }
__device__ __forceinline__ int edst(const int* ei, int e) { return e < E_RAW ? ei[E_RAW + e] : (e - E_RAW); }

// ---------------- tiny utility kernels ----------------
__global__ void k_zero32(unsigned* __restrict__ p, long n) {
  long i = (long)blockIdx.x * blockDim.x + threadIdx.x;
  if (i < n) p[i] = 0u;
}

__global__ void k_cvt_f16(const float* __restrict__ s, _Float16* __restrict__ d, long n) {
  long i = (long)blockIdx.x * blockDim.x + threadIdx.x;
  if (i < n) d[i] = (_Float16)s[i];
}

// transpose+convert weights: WT[n*K + k] = (f16) W[k*HC + n]   (W is [K x HC])
__global__ void k_cvt_wT(const float* __restrict__ W, _Float16* __restrict__ WT, int K) {
  long idx = (long)blockIdx.x * blockDim.x + threadIdx.x;
  if (idx >= (long)K * HC) return;
  int n = (int)(idx / K), k = (int)(idx % K);
  WT[idx] = (_Float16)W[(size_t)k * HC + n];
}

__global__ void k_add_bias(float* __restrict__ o, const float* __restrict__ b, long n) {
  long i = (long)blockIdx.x * blockDim.x + threadIdx.x;
  if (i < n) o[i] += b[i % HC];
}

// ---------------- WMMA GEMM: Y[M x Nout] = A[M x K](f16) @ WT[Nout x K](f16)^T ----------------
#define BM 128
#define BN 128
#define BK 32
#define LDS_STRIDE 40   // halves; 32 data + 8 pad -> 80B rows, 16B aligned, bank-spread

#if USE_TDM
// Issue one TDM 2D tile load: tile_dim0=BK (K, contiguous), tile_dim1=BM rows,
// 2-byte elements, LDS padding 4 DWORDs every 16 DWORDs (32->40 half stride),
// rows beyond rows_remaining zero-filled by TDM OOB handling.
__device__ __forceinline__ void tdm_load_2d(const _Float16* gptr, unsigned lds_byte,
                                            int rows_remaining, int ld_elems) {
  unsigned long long ga = (unsigned long long)(size_t)gptr;
  v4u g0;
  g0[0] = 1u;                                            // count=1, user descriptor
  g0[1] = lds_byte;                                      // lds_addr
  g0[2] = (unsigned)ga;                                  // global_addr[31:0]
  g0[3] = ((unsigned)(ga >> 32) & 0x01FFFFFFu) | (2u << 30); // addr[56:32] | type=2
  unsigned td0 = (unsigned)BK;                           // tensor_dim0 (>= tile_dim0, no X clip)
  unsigned td1 = (unsigned)rows_remaining;               // tensor_dim1 -> tail rows zero-filled
  unsigned ld  = (unsigned)ld_elems;                     // tensor_dim0_stride
  v8i g1;
  g1[0] = (int)((1u << 16)        // data_size = 1 -> 2 bytes
              | (1u << 20)        // pad_enable
              | (3u << 22)        // pad_interval = 3 -> every 16 DWORDs (one 32-half row)
              | (3u << 25));      // pad_amount   = 3 -> 4 DWORDs (8 halves)
  g1[1] = (int)(td0 << 16);                              // [63:48] = tensor_dim0 lo16
  g1[2] = (int)((td0 >> 16) | (td1 << 16));              // [79:64] hi16 | [95:80] dim1 lo16
  g1[3] = (int)((td1 >> 16) | ((unsigned)BK << 16));     // [111:96] hi16 | [127:112] tile_dim0
  g1[4] = (int)BM;                                       // [143:128] tile_dim1 ; tile_dim2=0
  g1[5] = (int)ld;                                       // [191:160] tensor_dim0_stride lo32
  g1[6] = 0;                                             // stride hi16, dim1_stride lo16
  g1[7] = 0;
  v4i zz = {0, 0, 0, 0};
#if __clang_major__ >= 23
  v8i z8 = {0, 0, 0, 0, 0, 0, 0, 0};
  __builtin_amdgcn_tensor_load_to_lds(g0, g1, zz, zz, z8, 0);
#else
  __builtin_amdgcn_tensor_load_to_lds(g0, g1, zz, zz, 0);
#endif
}
#endif // USE_TDM

__global__ __launch_bounds__(256)
void k_gemm_wmma(const _Float16* __restrict__ A, const _Float16* __restrict__ WT,
                 float* __restrict__ Y, int M, int K, int Nout) {
  constexpr int TILEH = BM * LDS_STRIDE;      // halves per tile (A or B)
  constexpr int STAGE = 2 * TILEH;            // A tile + B tile per buffer
  __shared__ __align__(16) _Float16 smem[2 * STAGE];   // double buffered: 40 KB

  const int tid  = threadIdx.x;
  const int lane = tid & 31;
  const int wave = tid >> 5;      // 0..7
  const int wm   = wave & 3;      // M group (32 rows each)
  const int wn   = wave >> 2;     // N group (64 cols each)
  const int rowBase = blockIdx.y * BM;
  const int colBase = blockIdx.x * BN;
  const int lrow = lane & 15;
  const int hsel = lane >> 4;     // 0: lanes 0-15, 1: lanes 16-31
  const int nk = K / BK;

  v8f acc[2][4];
#pragma unroll
  for (int mi = 0; mi < 2; ++mi)
#pragma unroll
    for (int ni = 0; ni < 4; ++ni)
#pragma unroll
      for (int j = 0; j < 8; ++j) acc[mi][ni][j] = 0.0f;

  // ---- staging: TDM DMA (preferred) or synchronous fallback ----
  auto stage = [&](int kt, int buf) {
#if USE_TDM
    if (wave == 0) {
      unsigned base = (unsigned)(size_t)&smem[0] + (unsigned)(buf * STAGE) * 2u;
      tdm_load_2d(A  + (size_t)rowBase * K + (size_t)kt * BK, base,             M    - rowBase, K);
      tdm_load_2d(WT + (size_t)colBase * K + (size_t)kt * BK, base + TILEH * 2, Nout - colBase, K);
    }
#else
    _Float16* sA = &smem[buf * STAGE];
    _Float16* sB = &smem[buf * STAGE + TILEH];
    const int k0 = kt * BK;
#pragma unroll
    for (int it = 0; it < 2; ++it) {
      int idx = it * 256 + tid;          // 0..511
      int r   = idx >> 2;                // 0..127
      int kc  = (idx & 3) << 3;          // 0,8,16,24
      int ga  = rowBase + r; if (ga >= M) ga = M - 1;   // clamp tail (stores guarded)
      *(v8h*)(&sA[r * LDS_STRIDE + kc]) = *(const v8h*)(A + (size_t)ga * K + k0 + kc);
      *(v8h*)(&sB[r * LDS_STRIDE + kc]) = *(const v8h*)(WT + (size_t)(colBase + r) * K + k0 + kc);
    }
#endif
  };
  auto wait_stage = [&]() {
#if USE_TDM
    if (wave == 0) {
#if __has_builtin(__builtin_amdgcn_s_wait_tensorcnt)
      __builtin_amdgcn_s_wait_tensorcnt(0);
#else
      asm volatile("s_wait_tensorcnt 0x0" ::: "memory");
#endif
    }
#endif
    __syncthreads();
  };

  stage(0, 0);
  for (int kt = 0; kt < nk; ++kt) {
    const int cur = kt & 1;
    wait_stage();                          // stage(kt) visible to all waves; prior reads done
    if (kt + 1 < nk) stage(kt + 1, cur ^ 1);  // overlap DMA of next tile with WMMAs

    const _Float16* sA = &smem[cur * STAGE];
    const _Float16* sB = &smem[cur * STAGE + TILEH];

    // A 16x32 fragment: lanes 0-15 -> a[0..7]=K0..7, a[8..15]=K16..23 ; lanes 16-31 -> +8
    v16h afr[2];
#pragma unroll
    for (int mi = 0; mi < 2; ++mi) {
      int r = wm * 32 + mi * 16 + lrow;
      v8h lo = *(const v8h*)(&sA[r * LDS_STRIDE + hsel * 8]);
      v8h hi = *(const v8h*)(&sA[r * LDS_STRIDE + 16 + hsel * 8]);
      afr[mi] = __builtin_shufflevector(lo, hi, 0,1,2,3,4,5,6,7,8,9,10,11,12,13,14,15);
    }
    // B 32x16 fragment: lane n = l&15 ; lanes 0-15 -> b[0..7]=K0..7, b[8..15]=K16..23 ; lanes 16-31 -> +8
    v16h bfr[4];
#pragma unroll
    for (int ni = 0; ni < 4; ++ni) {
      int n = wn * 64 + ni * 16 + lrow;
      v8h lo = *(const v8h*)(&sB[n * LDS_STRIDE + hsel * 8]);
      v8h hi = *(const v8h*)(&sB[n * LDS_STRIDE + 16 + hsel * 8]);
      bfr[ni] = __builtin_shufflevector(lo, hi, 0,1,2,3,4,5,6,7,8,9,10,11,12,13,14,15);
    }
#pragma unroll
    for (int mi = 0; mi < 2; ++mi)
#pragma unroll
      for (int ni = 0; ni < 4; ++ni)
        acc[mi][ni] = __builtin_amdgcn_wmma_f32_16x16x32_f16(
            false, afr[mi], false, bfr[ni], (short)0, acc[mi][ni], false, false);
  }

  // store: C/D layout VGPR r -> M=r (lanes 0-15) / M=r+8 (lanes 16-31)
#pragma unroll
  for (int mi = 0; mi < 2; ++mi)
#pragma unroll
    for (int ni = 0; ni < 4; ++ni) {
      int n  = colBase + wn * 64 + ni * 16 + lrow;
      int rb = rowBase + wm * 32 + mi * 16 + hsel * 8;
#pragma unroll
      for (int r = 0; r < 8; ++r) {
        int row = rb + r;
        if (row < M) Y[(size_t)row * Nout + n] = acc[mi][ni][r];
      }
    }
}

// ---------------- edge phase ----------------
__global__ void k_edge_logits(const float* __restrict__ xl, const float* __restrict__ xr,
                              const int* __restrict__ ei, const float* __restrict__ att,
                              float* __restrict__ logits, unsigned* __restrict__ maxb) {
  int idx = blockIdx.x * blockDim.x + threadIdx.x;
  if (idx >= NEDGE * NH) return;
  int e = idx >> 3, h = idx & 7;
  int s = esrc(ei, e), d = edst(ei, e);
  const float4* pl = (const float4*)(xl + (size_t)s * HC + h * NC);
  const float4* pr = (const float4*)(xr + (size_t)d * HC + h * NC);
  const float4* pa = (const float4*)(att + h * NC);
  float acc = 0.0f;
#pragma unroll
  for (int i = 0; i < NC / 4; ++i) {
    float4 a = pl[i], b = pr[i], w = pa[i];
    float v;
    v = a.x + b.x; v = v > 0.0f ? v : NEG_SLOPE * v; acc += v * w.x;
    v = a.y + b.y; v = v > 0.0f ? v : NEG_SLOPE * v; acc += v * w.y;
    v = a.z + b.z; v = v > 0.0f ? v : NEG_SLOPE * v; acc += v * w.z;
    v = a.w + b.w; v = v > 0.0f ? v : NEG_SLOPE * v; acc += v * w.w;
  }
  logits[idx] = acc;
  atomicMax(maxb + (size_t)d * NH + h, enc_f32(acc));
}

__global__ void k_edge_exp(float* __restrict__ logits, const unsigned* __restrict__ maxb,
                           float* __restrict__ sumb, const int* __restrict__ ei) {
  int idx = blockIdx.x * blockDim.x + threadIdx.x;
  if (idx >= NEDGE * NH) return;
  int e = idx >> 3, h = idx & 7;
  int d = edst(ei, e);
  float m = dec_f32(maxb[(size_t)d * NH + h]);   // finite: self-loops guarantee >=1 edge
  float ex = expf(logits[idx] - m);
  logits[idx] = ex;
  atomicAdd(sumb + (size_t)d * NH + h, ex);
}

__global__ __launch_bounds__(128)
void k_edge_aggr(const float* __restrict__ xl, const float* __restrict__ logits,
                 const float* __restrict__ sumb, const int* __restrict__ ei,
                 float* __restrict__ outb) {
  int e = blockIdx.x;
  int t = threadIdx.x;           // 128 threads * 4 ch = 512
  int s = esrc(ei, e), d = edst(ei, e);
  int j = t << 2;
  int h = j >> 6;
  float alpha = logits[(size_t)e * NH + h] / (sumb[(size_t)d * NH + h] + 1e-16f);
  float4 v = *(const float4*)(xl + (size_t)s * HC + j);
  float* o = outb + (size_t)d * HC + j;
  atomicAdd(o + 0, v.x * alpha);
  atomicAdd(o + 1, v.y * alpha);
  atomicAdd(o + 2, v.z * alpha);
  atomicAdd(o + 3, v.w * alpha);
}

// ---------------- post-GNN ----------------
__global__ void k_head_mean(const float* __restrict__ outb, const float* __restrict__ b2,
                            float* __restrict__ h3, long n) {
  long idx = (long)blockIdx.x * blockDim.x + threadIdx.x;
  if (idx >= n) return;
  long node = idx >> 6; int c = (int)(idx & 63);
  float s = 0.0f;
#pragma unroll
  for (int h = 0; h < NH; ++h) s += outb[node * HC + h * NC + c];
  h3[idx] = s * (1.0f / NH) + b2[c];
}

__global__ void k_node_w(const float* __restrict__ h3, const float* __restrict__ w_aw,
                         const float* __restrict__ b_aw, float* __restrict__ wn) {
  int n = blockIdx.x * blockDim.x + threadIdx.x;
  if (n >= NN) return;
  const float4* p = (const float4*)(h3 + (size_t)n * NC);
  const float4* w = (const float4*)w_aw;
  float acc = 0.0f;
#pragma unroll
  for (int i = 0; i < NC / 4; ++i) {
    float4 a = p[i], b = w[i];
    acc += a.x * b.x + a.y * b.y + a.z * b.z + a.w * b.w;
  }
  float t = acc + b_aw[0];
  wn[n] = 1.0f / (1.0f + expf(-t));
}

__global__ void k_pool(const float* __restrict__ h3, const float* __restrict__ wn,
                       const int* __restrict__ batch, unsigned* __restrict__ pmax,
                       float* __restrict__ psum) {
  long idx = (long)blockIdx.x * blockDim.x + threadIdx.x;
  if (idx >= (long)NN * NC) return;
  long node = idx >> 6; int c = (int)(idx & 63);
  int b = batch[node];
  float v = h3[idx];
  atomicMax(pmax + (size_t)b * NC + c, enc_f32(v));
  atomicAdd(psum + (size_t)b * NC + c, wn[node] * v);
}

__global__ __launch_bounds__(128)
void k_mlp(const unsigned* __restrict__ pmax, const float* __restrict__ psum,
           const float* __restrict__ Wm1, const float* __restrict__ bm1,
           const float* __restrict__ ap, const float* __restrict__ Wm2,
           const float* __restrict__ bm2, float* __restrict__ out) {
  __shared__ float g[2 * NC];
  __shared__ float red[2 * NC];
  int b = blockIdx.x, t = threadIdx.x;
  if (t < NC) {
    unsigned k = pmax[(size_t)b * NC + t];
    g[t] = (k == 0u) ? 0.0f : dec_f32(k);   // empty graph -> isfinite guard -> 0
  } else {
    g[t] = psum[(size_t)b * NC + (t - NC)];
  }
  __syncthreads();
  float z = bm1[t];
#pragma unroll 8
  for (int k = 0; k < 2 * NC; ++k) z += g[k] * Wm1[k * 128 + t];
  float a = ap[0];
  z = z > 0.0f ? z : a * z;                 // PReLU
  red[t] = z * Wm2[t];
  __syncthreads();
  for (int s = 64; s > 0; s >>= 1) {
    if (t < s) red[t] += red[t + s];
    __syncthreads();
  }
  if (t == 0) out[b] = red[0] + bm2[0];
}

// ---------------- host launcher ----------------
extern "C" void kernel_launch(void* const* d_in, const int* in_sizes, int n_in,
                              void* d_out, int out_size, void* d_ws, size_t ws_size,
                              hipStream_t stream) {
  (void)in_sizes; (void)n_in; (void)out_size; (void)ws_size;
  const float* x     = (const float*)d_in[0];
  const int*   ei    = (const int*)d_in[1];
  const int*   batch = (const int*)d_in[2];
  const float* Wl[3]   = {(const float*)d_in[3], (const float*)d_in[7],  (const float*)d_in[11]};
  const float* Wr[3]   = {(const float*)d_in[4], (const float*)d_in[8],  (const float*)d_in[12]};
  const float* att[3]  = {(const float*)d_in[5], (const float*)d_in[9],  (const float*)d_in[13]};
  const float* bias[3] = {(const float*)d_in[6], (const float*)d_in[10], (const float*)d_in[14]};
  const float* w_aw = (const float*)d_in[15];
  const float* b_aw = (const float*)d_in[16];
  const float* Wm1  = (const float*)d_in[17];
  const float* bm1  = (const float*)d_in[18];
  const float* ap   = (const float*)d_in[19];
  const float* Wm2  = (const float*)d_in[20];
  const float* bm2  = (const float*)d_in[21];
  float* out = (float*)d_out;

  char* ws = (char*)d_ws;
  size_t off = 0;
  auto take = [&](size_t bytes) -> char* {
    char* p = ws + off;
    off += (bytes + 255) & ~(size_t)255;
    return p;
  };
  _Float16* A16   = (_Float16*)take((size_t)NN * HC * 2);
  _Float16* WLT16 = (_Float16*)take((size_t)HC * HC * 2);  // transposed [Nout x K]
  _Float16* WRT16 = (_Float16*)take((size_t)HC * HC * 2);
  float*    XL   = (float*)take((size_t)NN * HC * 4);
  float*    XR   = (float*)take((size_t)NN * HC * 4);
  float*    OUTB = XR;   // alias: XR dead after k_edge_logits; OUT consumed before next gemm
  float*    LOG  = (float*)take((size_t)NEDGE * NH * 4);
  unsigned* MAXB = (unsigned*)take((size_t)NN * NH * 4);
  float*    SUMB = (float*)take((size_t)NN * NH * 4);
  float*    H3   = (float*)take((size_t)NN * NC * 4);
  float*    WN   = (float*)take((size_t)NN * 4);
  unsigned* PMAX = (unsigned*)take((size_t)NBATCH * NC * 4);
  float*    PSUM = (float*)take((size_t)NBATCH * NC * 4);

  auto blks = [](long n, int b) { return (unsigned)((n + b - 1) / b); };

  const float* Xin = x;
  int K = F0;
  for (int L = 0; L < 3; ++L) {
    long nx = (long)NN * K;
    k_cvt_f16<<<blks(nx, 256), 256, 0, stream>>>(Xin, A16, nx);
    long nw = (long)K * HC;
    k_cvt_wT<<<blks(nw, 256), 256, 0, stream>>>(Wl[L], WLT16, K);
    k_cvt_wT<<<blks(nw, 256), 256, 0, stream>>>(Wr[L], WRT16, K);

    dim3 gg(HC / BN, (NN + BM - 1) / BM);
    k_gemm_wmma<<<gg, 256, 0, stream>>>(A16, WLT16, XL, NN, K, HC);
    k_gemm_wmma<<<gg, 256, 0, stream>>>(A16, WRT16, XR, NN, K, HC);

    long nm = (long)NN * NH;
    k_zero32<<<blks(nm, 256), 256, 0, stream>>>(MAXB, nm);
    k_zero32<<<blks(nm, 256), 256, 0, stream>>>((unsigned*)SUMB, nm);

    long nl = (long)NEDGE * NH;
    k_edge_logits<<<blks(nl, 256), 256, 0, stream>>>(XL, XR, ei, att[L], LOG, MAXB);
    k_edge_exp<<<blks(nl, 256), 256, 0, stream>>>(LOG, MAXB, SUMB, ei);

    long no = (long)NN * HC;
    k_zero32<<<blks(no, 256), 256, 0, stream>>>((unsigned*)OUTB, no);  // XR dead now
    k_edge_aggr<<<NEDGE, 128, 0, stream>>>(XL, LOG, SUMB, ei, OUTB);

    if (L < 2) {
      k_add_bias<<<blks(no, 256), 256, 0, stream>>>(OUTB, bias[L], no);
      Xin = OUTB;
      K = HC;
    } else {
      long nh3 = (long)NN * NC;
      k_head_mean<<<blks(nh3, 256), 256, 0, stream>>>(OUTB, bias[2], H3, nh3);
    }
  }

  k_node_w<<<blks(NN, 256), 256, 0, stream>>>(H3, w_aw, b_aw, WN);
  long np = (long)NBATCH * NC;
  k_zero32<<<blks(np, 256), 256, 0, stream>>>(PMAX, np);
  k_zero32<<<blks(np, 256), 256, 0, stream>>>((unsigned*)PSUM, np);
  long nn = (long)NN * NC;
  k_pool<<<blks(nn, 256), 256, 0, stream>>>(H3, WN, batch, PMAX, PSUM);
  k_mlp<<<NBATCH, 128, 0, stream>>>(PMAX, PSUM, Wm1, bm1, ap, Wm2, bm2, out);
}
